// CisAttentionLayer_16355235463522
// MI455X (gfx1250) — matrix operand order repaired and checked by reading, not compiled, not run on hardware
//
#include <hip/hip_runtime.h>
#include <hip/hip_bf16.h>

#define B_  4
#define G_  1024
#define S_  4096
#define D_  512
#define H_  8
#define DK_ 64

#define AS1 __attribute__((address_space(1)))
#define AS3 __attribute__((address_space(3)))

typedef __attribute__((ext_vector_type(16))) __bf16 v16bf;
typedef __attribute__((ext_vector_type(8)))  __bf16 v8bf;
typedef __attribute__((ext_vector_type(4)))  __bf16 v4bf;
typedef __attribute__((ext_vector_type(8)))  float  v8f;
typedef __attribute__((ext_vector_type(4)))  float  v4f;
typedef __attribute__((ext_vector_type(4)))  int    v4i;

// ---- CDNA5 async global->LDS copy (ASYNCcnt) with inline-asm fallback ----
__device__ __forceinline__ void async_cp_b128(void* lds_dst, const void* gsrc) {
#if __has_builtin(__builtin_amdgcn_global_load_async_to_lds_b128)
  __builtin_amdgcn_global_load_async_to_lds_b128(
      (AS1 v4i*)(uintptr_t)gsrc,
      (AS3 v4i*)(uint32_t)(uintptr_t)lds_dst, 0, 0);
#else
  asm volatile("global_load_async_to_lds_b128 %0, %1, off"
               :: "v"((uint32_t)(uintptr_t)lds_dst),
                  "v"((uint64_t)(uintptr_t)gsrc)
               : "memory");
#endif
}

__device__ __forceinline__ void wait_async_le2() {
#if __has_builtin(__builtin_amdgcn_s_wait_asynccnt)
  __builtin_amdgcn_s_wait_asynccnt(2);
#else
  asm volatile("s_wait_asynccnt 2" ::: "memory");
#endif
}
__device__ __forceinline__ void wait_async_0() {
#if __has_builtin(__builtin_amdgcn_s_wait_asynccnt)
  __builtin_amdgcn_s_wait_asynccnt(0);
#else
  asm volatile("s_wait_asynccnt 0" ::: "memory");
#endif
}

__device__ __forceinline__ v16bf cat16(v8bf lo, v8bf hi) {
  v16bf r;
#pragma unroll
  for (int i = 0; i < 8; ++i) { r[i] = lo[i]; r[i + 8] = hi[i]; }
  return r;
}

__device__ __forceinline__ v16bf cvt16(v4f a0, v4f a1, v4f a2, v4f a3) {
  v16bf r;
#pragma unroll
  for (int i = 0; i < 4; ++i) {
    r[i]      = (__bf16)a0[i];
    r[4 + i]  = (__bf16)a1[i];
    r[8 + i]  = (__bf16)a2[i];
    r[12 + i] = (__bf16)a3[i];
  }
  return r;
}

// ---------------- fp32 -> bf16 weight conversion ----------------
__global__ __launch_bounds__(256) void cvt_bf16_kernel(const float* __restrict__ src,
                                                       __bf16* __restrict__ dst, int n) {
  int i = (blockIdx.x * blockDim.x + threadIdx.x) * 4;
  if (i < n) {
    v4f v = *(const v4f*)(src + i);
    v4bf o;
#pragma unroll
    for (int k = 0; k < 4; ++k) o[k] = (__bf16)v[k];
    *(v4bf*)(dst + i) = o;
  }
}

// ---------------- input projection: Y = X @ W^T + bias (+shift), bf16 WMMA ----------------
// vmode 0: store row-major bf16 [B,rows,D]   (Q with 1/sqrt(dk) folded in, K)
// vmode 1: store head-transposed Vt[b,h,d,s] (V) as packed 8-wide bf16 stores
__global__ __launch_bounds__(256) void proj_kernel(
    const float* __restrict__ X, const __bf16* __restrict__ W,
    const float* __restrict__ bias, const float* __restrict__ shift,
    __bf16* __restrict__ dst, int rows, float scale, int vmode)
{
  const int lane = threadIdx.x & 31;
  const int half = lane >> 4;
  const int l    = lane & 15;
  const int w    = (blockIdx.x * blockDim.x + threadIdx.x) >> 5;
  const int ntiles = D_ / 16;
  const int mtiles = rows / 16;
  const int b  = w / (mtiles * ntiles);
  const int rm = w - b * (mtiles * ntiles);
  const int mt = rm / ntiles;
  const int nt = rm - mt * ntiles;

  const float*  xrow = X + ((size_t)(b * rows + mt * 16 + l)) * D_;
  const __bf16* wrow = W + ((size_t)(nt * 16 + l)) * D_ + 8 * half;

  v8f c = {};
#pragma unroll 4
  for (int k0 = 0; k0 < D_; k0 += 32) {
    const int c1 = k0 + 8 * half;
    v4f a0 = *(const v4f*)(xrow + c1);
    v4f a1 = *(const v4f*)(xrow + c1 + 4);
    v4f a2 = *(const v4f*)(xrow + c1 + 16);
    v4f a3 = *(const v4f*)(xrow + c1 + 20);
    v16bf a = cvt16(a0, a1, a2, a3);
    v8bf b0 = *(const v8bf*)(wrow + k0);
    v8bf b1 = *(const v8bf*)(wrow + k0 + 16);
    v16bf bb = cat16(b0, b1);
    c = __builtin_amdgcn_wmma_f32_16x16x32_bf16(false, a, false, bb, (short)0, c, false, false);
  }

  const int n = nt * 16 + l;
  const float add = bias[n] + (shift ? shift[b * D_ + n] : 0.f);

  if (vmode == 0) {
#pragma unroll
    for (int r = 0; r < 8; ++r) {
      const int m = mt * 16 + r + 8 * half;
      dst[((size_t)(b * rows + m)) * D_ + n] = (__bf16)((c[r] + add) * scale);
    }
  } else {
    const int h = n >> 6, d = n & 63;
    v8bf pk;
#pragma unroll
    for (int r = 0; r < 8; ++r) pk[r] = (__bf16)(c[r] + add);
    __bf16* p = dst + ((size_t)((b * H_ + h) * DK_ + d)) * S_ + mt * 16 + 8 * half;
    *(v8bf*)p = pk;   // 8 consecutive s values for one d column
  }
}

// ---------------- flash attention ----------------
// Block = 8 waves sharing one (b,h); K/V tiles double-buffered in LDS via
// async global->LDS copies (ASYNCcnt), each wave owns one 16-query tile.
__global__ __launch_bounds__(256) void attn_kernel(
    const __bf16* __restrict__ Qb, const __bf16* __restrict__ Kb,
    const __bf16* __restrict__ Vt, const int* __restrict__ mask,
    const unsigned char* __restrict__ cis, __bf16* __restrict__ O)
{
  __shared__ __attribute__((aligned(16))) __bf16 KT[2][32 * 64];    // [s][d]
  __shared__ __attribute__((aligned(16))) __bf16 VTile[2][64 * 32]; // [d][s]
  __shared__ float Pl[8][16 * 32];

  const int tid  = threadIdx.x;
  const int lane = tid & 31, half = lane >> 4, l = lane & 15;
  const int wv   = tid >> 5;
  const int bh   = blockIdx.x >> 3;          // 0..31 : (b,h)
  const int gtg  = blockIdx.x & 7;           // group of 8 query tiles
  const int b = bh >> 3, h = bh & 7;
  const int gt = gtg * 8 + wv;

  const __bf16* Kbase = Kb + (size_t)(b * S_) * D_ + h * DK_;
  const __bf16* Vbase = Vt + (size_t)((b * H_ + h) * DK_) * S_;

  // staging roles: K tile 32 rows x (8 segs of 8 bf16); V tile 64 rows x 4 segs
  const int krow = tid >> 3, kseg = tid & 7;
  const int vrow = tid >> 2, vseg = tid & 3;

  auto stage = [&](int buf, int s0) {
    async_cp_b128(&KT[buf][krow * 64 + kseg * 8],
                  Kbase + (size_t)(s0 + krow) * D_ + kseg * 8);
    async_cp_b128(&VTile[buf][vrow * 32 + vseg * 8],
                  Vbase + (size_t)vrow * S_ + s0 + vseg * 8);
  };

  stage(0, 0);

  // Q fragments for d=0..63 (scale folded into Qb)
  v16bf qa[2];
  {
    const __bf16* qrow = Qb + ((size_t)(b * G_ + gt * 16 + l)) * D_ + h * DK_;
#pragma unroll
    for (int j = 0; j < 2; ++j) {
      const int c1 = 32 * j + 8 * half;
      v8bf q0 = *(const v8bf*)(qrow + c1);
      v8bf q1 = *(const v8bf*)(qrow + c1 + 16);
      qa[j] = cat16(q0, q1);
    }
  }

  float m_i[8], l_i[8];
  v8f acc[4];
#pragma unroll
  for (int r = 0; r < 8; ++r) { m_i[r] = -3.0e38f; l_i[r] = 0.f; }
#pragma unroll
  for (int dt = 0; dt < 4; ++dt) acc[dt] = (v8f){};

  float* pl = &Pl[wv][0];

  for (int s0 = 0; s0 < S_; s0 += 32) {
    const int buf = (s0 >> 5) & 1;
    if (s0 + 32 < S_) { stage(buf ^ 1, s0 + 32); wait_async_le2(); }
    else              { wait_async_0(); }
    __syncthreads();   // staged K/V visible to all waves

    float sc[2][8], mnew[8];
#pragma unroll
    for (int r = 0; r < 8; ++r) mnew[r] = m_i[r];

#pragma unroll
    for (int t = 0; t < 2; ++t) {
      v8f c = {};
#pragma unroll
      for (int dj = 0; dj < 2; ++dj) {
        const __bf16* kr = &KT[buf][(16 * t + l) * 64 + dj * 32 + 8 * half];
        v8bf k0v = *(const v8bf*)kr;
        v8bf k1v = *(const v8bf*)(kr + 16);
        v16bf kb = cat16(k0v, k1v);
        c = __builtin_amdgcn_wmma_f32_16x16x32_bf16(false, qa[dj], false, kb, (short)0, c, false, false);
      }
      const int scol = s0 + 16 * t + l;
      const int pm   = mask[b * S_ + scol];
      const unsigned char* cp = cis + (size_t)(gt * 16 + 8 * half) * S_ + scol;
#pragma unroll
      for (int r = 0; r < 8; ++r) {
        const bool ok = (pm != 0) && (cp[(size_t)r * S_] != 0);
        sc[t][r] = ok ? c[r] : -1.0e9f;
        mnew[r]  = fmaxf(mnew[r], sc[t][r]);
      }
    }

    // row-max across the 16 lanes of each half
#pragma unroll
    for (int off = 1; off < 16; off <<= 1)
#pragma unroll
      for (int r = 0; r < 8; ++r)
        mnew[r] = fmaxf(mnew[r], __shfl_xor(mnew[r], off, 32));

    float p[2][8], psum[8];
#pragma unroll
    for (int r = 0; r < 8; ++r) {
      p[0][r] = __expf(sc[0][r] - mnew[r]);
      p[1][r] = __expf(sc[1][r] - mnew[r]);
      psum[r] = p[0][r] + p[1][r];
    }
#pragma unroll
    for (int off = 1; off < 16; off <<= 1)
#pragma unroll
      for (int r = 0; r < 8; ++r)
        psum[r] += __shfl_xor(psum[r], off, 32);

#pragma unroll
    for (int r = 0; r < 8; ++r) {
      const float corr = __expf(m_i[r] - mnew[r]);
      l_i[r] = l_i[r] * corr + psum[r];
      m_i[r] = mnew[r];
#pragma unroll
      for (int dt = 0; dt < 4; ++dt) acc[dt][r] *= corr;
    }

    // stage P through (wave-private) LDS: C-fragment layout -> A-fragment layout
#pragma unroll
    for (int t = 0; t < 2; ++t)
#pragma unroll
      for (int r = 0; r < 8; ++r)
        pl[(r + 8 * half) * 32 + 16 * t + l] = p[t][r];

    v16bf pa;
    {
      const int c1 = 8 * half, c2 = 16 + 8 * half;
#pragma unroll
      for (int e = 0; e < 8; ++e) {
        pa[e]     = (__bf16)pl[l * 32 + c1 + e];
        pa[e + 8] = (__bf16)pl[l * 32 + c2 + e];
      }
    }

#pragma unroll
    for (int dt = 0; dt < 4; ++dt) {
      const __bf16* vr = &VTile[buf][(dt * 16 + l) * 32 + 8 * half];
      v8bf v0 = *(const v8bf*)vr;
      v8bf v1 = *(const v8bf*)(vr + 16);
      v16bf vb = cat16(v0, v1);
      acc[dt] = __builtin_amdgcn_wmma_f32_16x16x32_bf16(false, pa, false, vb, (short)0, acc[dt], false, false);
    }

    __syncthreads();   // all reads of this buffer done before it is re-staged
  }

#pragma unroll
  for (int r = 0; r < 8; ++r) {
    const float inv = 1.f / l_i[r];
    const int g = gt * 16 + r + 8 * half;
#pragma unroll
    for (int dt = 0; dt < 4; ++dt)
      O[((size_t)(b * G_ + g)) * D_ + h * DK_ + dt * 16 + l] = (__bf16)(acc[dt][r] * inv);
  }
}

// ---------------- output projection: out2 = O @ Wo^T + wo_b (f32 out) ----------------
__global__ __launch_bounds__(256) void oproj_kernel(
    const __bf16* __restrict__ Xb, const __bf16* __restrict__ W,
    const float* __restrict__ bias, float* __restrict__ dst)
{
  const int lane = threadIdx.x & 31, half = lane >> 4, l = lane & 15;
  const int w = (blockIdx.x * blockDim.x + threadIdx.x) >> 5;
  const int ntiles = D_ / 16;
  const int mt = w / ntiles;
  const int nt = w - mt * ntiles;
  const __bf16* xrow = Xb + ((size_t)(mt * 16 + l)) * D_;
  const __bf16* wrow = W + ((size_t)(nt * 16 + l)) * D_ + 8 * half;
  v8f c = {};
#pragma unroll 4
  for (int k0 = 0; k0 < D_; k0 += 32) {
    const int c1 = k0 + 8 * half;
    v8bf a0 = *(const v8bf*)(xrow + c1);
    v8bf a1 = *(const v8bf*)(xrow + c1 + 16);
    v16bf a = cat16(a0, a1);
    v8bf b0 = *(const v8bf*)(wrow + k0);
    v8bf b1 = *(const v8bf*)(wrow + k0 + 16);
    v16bf bb = cat16(b0, b1);
    c = __builtin_amdgcn_wmma_f32_16x16x32_bf16(false, a, false, bb, (short)0, c, false, false);
  }
  const int n = nt * 16 + l;
  const float bv = bias[n];
#pragma unroll
  for (int r = 0; r < 8; ++r) {
    const int m = mt * 16 + r + 8 * half;
    dst[(size_t)m * D_ + n] = c[r] + bv;
  }
}

// ---------------- row-wise layernorm over D=512 ----------------
__global__ __launch_bounds__(128) void ln_kernel(const float* __restrict__ x,
                                                 const float* __restrict__ gam,
                                                 const float* __restrict__ bet,
                                                 float* __restrict__ out) {
  const int row = blockIdx.x;
  const int t = threadIdx.x;                // 128 threads, 4 elems each
  const float* xr = x + (size_t)row * D_;
  v4f v = *(const v4f*)(xr + t * 4);
  float s = v[0] + v[1] + v[2] + v[3];
  float q = v[0] * v[0] + v[1] * v[1] + v[2] * v[2] + v[3] * v[3];
#pragma unroll
  for (int off = 1; off < 32; off <<= 1) {
    s += __shfl_xor(s, off, 32);
    q += __shfl_xor(q, off, 32);
  }
  __shared__ float sh[8];
  const int wv = t >> 5;
  if ((t & 31) == 0) { sh[wv] = s; sh[4 + wv] = q; }
  __syncthreads();
  const float ts = sh[0] + sh[1] + sh[2] + sh[3];
  const float tq = sh[4] + sh[5] + sh[6] + sh[7];
  const float mu  = ts * (1.f / D_);
  const float var = tq * (1.f / D_) - mu * mu;
  const float rs  = rsqrtf(var + 1e-5f);
#pragma unroll
  for (int i = 0; i < 4; ++i) {
    const int cidx = t * 4 + i;
    out[(size_t)row * D_ + cidx] = (v[i] - mu) * rs * gam[cidx] + bet[cidx];
  }
}

extern "C" void kernel_launch(void* const* d_in, const int* in_sizes, int n_in,
                              void* d_out, int out_size, void* d_ws, size_t ws_size,
                              hipStream_t stream) {
  (void)in_sizes; (void)n_in; (void)out_size; (void)ws_size;
  const float* queries = (const float*)d_in[0];
  const float* kv      = (const float*)d_in[1];
  const float* dq      = (const float*)d_in[2];
  const float* dk      = (const float*)d_in[3];
  const int*   mask    = (const int*)d_in[4];
  const unsigned char* cis = (const unsigned char*)d_in[5];   // bool mask
  const float* wq_w = (const float*)d_in[6];
  const float* wq_b = (const float*)d_in[7];
  const float* wk_w = (const float*)d_in[8];
  const float* wk_b = (const float*)d_in[9];
  const float* wv_w = (const float*)d_in[10];
  const float* wv_b = (const float*)d_in[11];
  const float* wo_w = (const float*)d_in[12];
  const float* wo_b = (const float*)d_in[13];
  const float* ln_g = (const float*)d_in[14];
  const float* ln_b = (const float*)d_in[15];
  float* out = (float*)d_out;

  char* ws = (char*)d_ws;
  size_t off = 0;
  __bf16* Wq = (__bf16*)(ws + off); off += (size_t)D_ * D_ * 2;
  __bf16* Wk = (__bf16*)(ws + off); off += (size_t)D_ * D_ * 2;
  __bf16* Wv = (__bf16*)(ws + off); off += (size_t)D_ * D_ * 2;
  __bf16* Wo = (__bf16*)(ws + off); off += (size_t)D_ * D_ * 2;
  __bf16* Qb = (__bf16*)(ws + off); off += (size_t)B_ * G_ * D_ * 2;
  __bf16* Kb = (__bf16*)(ws + off); off += (size_t)B_ * S_ * D_ * 2;
  __bf16* Vt = (__bf16*)(ws + off); off += (size_t)B_ * S_ * D_ * 2;
  __bf16* O  = (__bf16*)(ws + off); off += (size_t)B_ * G_ * D_ * 2;
  float*  o2 = (float*)(ws + off);  off += (size_t)B_ * G_ * D_ * 4;

  const int WEL = D_ * D_;                     // 262144 elems per weight
  const int cvtBlocks = WEL / 4 / 256;         // 256
  cvt_bf16_kernel<<<cvtBlocks, 256, 0, stream>>>(wq_w, Wq, WEL);
  cvt_bf16_kernel<<<cvtBlocks, 256, 0, stream>>>(wk_w, Wk, WEL);
  cvt_bf16_kernel<<<cvtBlocks, 256, 0, stream>>>(wv_w, Wv, WEL);
  cvt_bf16_kernel<<<cvtBlocks, 256, 0, stream>>>(wo_w, Wo, WEL);

  // waves = B*(rows/16)*(D/16); 8 waves / 256-thread block
  proj_kernel<<<(B_ * (G_ / 16) * (D_ / 16)) / 8, 256, 0, stream>>>(
      queries, Wq, wq_b, dq, Qb, G_, 0.125f, 0);          // scale = 1/sqrt(64)
  proj_kernel<<<(B_ * (S_ / 16) * (D_ / 16)) / 8, 256, 0, stream>>>(
      kv, Wk, wk_b, dk, Kb, S_, 1.0f, 0);
  proj_kernel<<<(B_ * (S_ / 16) * (D_ / 16)) / 8, 256, 0, stream>>>(
      kv, Wv, wv_b, nullptr, Vt, S_, 1.0f, 1);

  attn_kernel<<<(B_ * H_ * (G_ / 16)) / 8, 256, 0, stream>>>(Qb, Kb, Vt, mask, cis, O);

  oproj_kernel<<<((B_ * G_ / 16) * (D_ / 16)) / 8, 256, 0, stream>>>(O, Wo, wo_b, o2);

  ln_kernel<<<B_ * G_, 128, 0, stream>>>(o2, ln_g, ln_b, out);
}
